// RNNDecoder_66048007078096
// MI455X (gfx1250) — compile-verified, matrix operand order checked by reading, not run on hardware
//
#include <hip/hip_runtime.h>

typedef __attribute__((ext_vector_type(16))) _Float16 v16h;
typedef __attribute__((ext_vector_type(8)))  float    v8f;

// B=64, T=512, H=1024
#define TT 512
#define HH 1024

static __device__ __forceinline__ float fast_sigmoid(float x) {
  return 1.0f / (1.0f + __expf(-x));
}
static __device__ __forceinline__ float fast_tanh(float x) {
  x = fminf(fmaxf(x, -15.0f), 15.0f);
  float e = __expf(2.0f * x);
  return (e - 1.0f) / (e + 1.0f);
}

// ---------------- kernel: zero the small accumulators in workspace ----------
__global__ void k_zero(float* ws) {
  for (int i = threadIdx.x; i < 3200; i += 256) ws[i] = 0.0f;
}

// ---------------- kernel: batch-norm statistics (sum / sumsq per channel) ---
__global__ __launch_bounds__(256) void k_stats(const float* __restrict__ x,
                                               float* __restrict__ sum,
                                               float* __restrict__ sumsq) {
  const int tid = threadIdx.x;
  const int row0 = blockIdx.x * 64;
  float s[4] = {0.f, 0.f, 0.f, 0.f}, q[4] = {0.f, 0.f, 0.f, 0.f};
  for (int rr = 0; rr < 64; ++rr) {
    const float* rp = x + (size_t)(row0 + rr) * HH;
#pragma unroll
    for (int cc = 0; cc < 4; ++cc) {
      float v = rp[tid + cc * 256];
      s[cc] += v;
      q[cc] += v * v;
    }
  }
#pragma unroll
  for (int cc = 0; cc < 4; ++cc) {
    atomicAdd(&sum[tid + cc * 256], s[cc]);
    atomicAdd(&sumsq[tid + cc * 256], q[cc]);
  }
}

// ---------------- kernel: fold BN affine + projection into cW[h], C --------
__global__ void k_params(const float* __restrict__ sum, const float* __restrict__ sumsq,
                         const float* __restrict__ gamma, const float* __restrict__ beta,
                         const float* __restrict__ Wp,
                         float* __restrict__ cW, float* __restrict__ Cacc) {
  int c = blockIdx.x * 256 + threadIdx.x;
  const float invN = 1.0f / 32768.0f;
  float mean = sum[c] * invN;
  float var = sumsq[c] * invN - mean * mean;
  float inv = rsqrtf(var + 1e-5f);
  float g = gamma[c] * inv;
  cW[c] = g * Wp[c];
  atomicAdd(Cacc, (beta[c] - mean * g) * Wp[c]);
}

// ---------------- kernel: px[b,t] = dot(x[b,t,:], cW) + C + bp -------------
__global__ __launch_bounds__(256) void k_px(const float* __restrict__ x,
                                            const float* __restrict__ cW,
                                            const float* __restrict__ Cacc,
                                            const float* __restrict__ bp,
                                            float* __restrict__ px) {
  const int lane = threadIdx.x & 31;
  const int wv = threadIdx.x >> 5;
  const int row = blockIdx.x * 8 + wv;
  const float* rp = x + (size_t)row * HH;
  float acc = 0.0f;
#pragma unroll 8
  for (int i = 0; i < 32; ++i) {
    int c = lane + i * 32;
    acc = fmaf(rp[c], cW[c], acc);
  }
#pragma unroll
  for (int off = 1; off < 32; off <<= 1) acc += __shfl_xor(acc, off, 32);
  if (lane == 0) px[row] = acc + Cacc[0] + bp[0];
}

// ---------------- kernel: pack W_ih1 (3072x1024 f32) into f16 WMMA B-frags --
// Fragment layout (16-bit operand, 16x16x32): lane n holds column n of the
// tile; within a lane, f16 index fi in [0,16) holds K per ISA table:
//   K in [0,8)   -> lanes 0-15,  fi = K
//   K in [8,16)  -> lanes 16-31, fi = K-8
//   K in [16,24) -> lanes 0-15,  fi = 8+(K-16)
//   K in [24,32) -> lanes 16-31, fi = 8+(K-24)
__global__ __launch_bounds__(256) void k_pack(const float* __restrict__ W,
                                              _Float16* __restrict__ wpack) {
  int gid = blockIdx.x * 256 + threadIdx.x;  // 0 .. 3072*1024-1
  int n = gid >> 10, k = gid & 1023;
  int nt = n >> 4, ln = n & 15;
  int kb = k >> 5, kr = k & 31;
  int grp = kr >> 3, kq = kr & 7;
  int lane = ln + ((grp & 1) << 4);
  int fi = kq + ((grp >> 1) << 3);
  size_t dst = (((size_t)(nt * 32 + kb)) * 32 + lane) * 16 + fi;
  wpack[dst] = (_Float16)W[gid];
}

// ---------------- kernel: the serial scan, one workgroup per 16 batches -----
#define H0_STRIDE 1032  // f16 stride; 516 dwords -> 16 distinct LDS banks for the A-gather

__global__ __launch_bounds__(256) void k_scan(
    const float* __restrict__ px,
    const float* __restrict__ W_ih0, const float* __restrict__ b_ih0,
    const float* __restrict__ b_hh0,
    const float* __restrict__ b_ih1, const float* __restrict__ b_hh1,
    const float* __restrict__ Wp, const float* __restrict__ bp,
    const _Float16* __restrict__ wpack,
    float* __restrict__ out) {
  __shared__ _Float16 h0s[16 * H0_STRIDE];  // 33.0 KB: layer-0 output, f16
  __shared__ float WrS[HH], WzS[HH], WnS[HH];
  __shared__ float BrS[HH], BzS[HH], BnS[HH], BhnS[HH];
  __shared__ float sS[16];     // per-batch scalar GRU input
  __shared__ float fpart[128]; // per-wave partial forecasts

  const int tid = threadIdx.x;
  const int lane = tid & 31;
  const int wave = tid >> 5;
  const int b0 = blockIdx.x * 16;
  const float bp0 = bp[0];

  // preload layer-0 combined parameters (h_prev==0 => W_hh0 unused)
  for (int j = tid; j < HH; j += 256) {
    WrS[j] = W_ih0[j];
    WzS[j] = W_ih0[HH + j];
    WnS[j] = W_ih0[2 * HH + j];
    BrS[j] = b_ih0[j] + b_hh0[j];
    BzS[j] = b_ih0[HH + j] + b_hh0[HH + j];
    BnS[j] = b_ih0[2 * HH + j];
    BhnS[j] = b_hh0[2 * HH + j];
  }
  if (tid < 16) sS[tid] = 0.5f * px[(size_t)(b0 + tid) * TT];  // forecast_0 = 0
  __syncthreads();

  const int mrow = lane & 15;
  const int khalf = (lane >> 4) << 3;  // 0 or 8
  const _Float16* arow = h0s + mrow * H0_STRIDE;

  for (int t = 0; t < TT; ++t) {
    // ---- layer-0 GRU cell, elementwise in s, write h0 tile to LDS as f16 ----
    for (int idx = tid; idx < 16 * HH; idx += 256) {
      int m = idx >> 10, j = idx & 1023;
      float s = sS[m];
      float r = fast_sigmoid(fmaf(s, WrS[j], BrS[j]));
      float z = fast_sigmoid(fmaf(s, WzS[j], BzS[j]));
      float nn = fast_tanh(fmaf(s, WnS[j], BnS[j]) + r * BhnS[j]);
      h0s[m * H0_STRIDE + j] = (_Float16)((1.0f - z) * nn);
    }
    __syncthreads();

    // ---- layer-1 GRU: [16,1024]x[1024,3072] via WMMA; wave owns cols
    //      [wave*128, wave*128+128) of each of the r/z/n gate blocks -------
    float part[8] = {0.f, 0.f, 0.f, 0.f, 0.f, 0.f, 0.f, 0.f};
    for (int j = 0; j < 8; ++j) {
      const int tR = wave * 8 + j;   // r-gate N-tile
      const int tZ = tR + 64;        // z-gate N-tile
      const int tN = tR + 128;       // n-gate N-tile
      if (j < 7) {  // warm GL2/WGP$ for next group's B fragments
        __builtin_prefetch(wpack + (((size_t)((tR + 1) * 32)) * 32 + lane) * 16, 0, 3);
        __builtin_prefetch(wpack + (((size_t)((tN + 1) * 32)) * 32 + lane) * 16, 0, 3);
      }
      v8f accR = {}, accZ = {}, accN = {};
      for (int kb = 0; kb < 32; ++kb) {
        // A fragment: gather 8 dwords (16 f16) from the LDS h0 tile
        union { v16h v; unsigned u[8]; } af;
        const int k0 = kb * 32 + khalf;
#pragma unroll
        for (int v = 0; v < 4; ++v) {
          af.u[v]     = *(const unsigned*)(arow + k0 + 2 * v);
          af.u[4 + v] = *(const unsigned*)(arow + k0 + 16 + 2 * v);
        }
        // B fragments: contiguous 32B per lane, pre-packed
        const v16h bR = *(const v16h*)(wpack + (((size_t)(tR * 32 + kb)) * 32 + lane) * 16);
        const v16h bZ = *(const v16h*)(wpack + (((size_t)(tZ * 32 + kb)) * 32 + lane) * 16);
        const v16h bN = *(const v16h*)(wpack + (((size_t)(tN * 32 + kb)) * 32 + lane) * 16);
        accR = __builtin_amdgcn_wmma_f32_16x16x32_f16(false, af.v, false, bR, (short)0, accR, false, false);
        accZ = __builtin_amdgcn_wmma_f32_16x16x32_f16(false, af.v, false, bZ, (short)0, accZ, false, false);
        accN = __builtin_amdgcn_wmma_f32_16x16x32_f16(false, af.v, false, bN, (short)0, accN, false, false);
      }
      // gate fusion: r/z/n accumulators are element-aligned in registers
      const int col = wave * 128 + j * 16 + mrow;  // hidden-unit index
      const float br = b_ih1[col] + b_hh1[col];
      const float bz = b_ih1[HH + col] + b_hh1[HH + col];
      const float bni = b_ih1[2 * HH + col];
      const float bnh = b_hh1[2 * HH + col];
      const float wpv = Wp[col];
#pragma unroll
      for (int v = 0; v < 8; ++v) {
        float r = fast_sigmoid(accR[v] + br);
        float z = fast_sigmoid(accZ[v] + bz);
        float nn = fast_tanh(accN[v] + bni + r * bnh);
        part[v] = fmaf((1.0f - z) * nn, wpv, part[v]);
      }
    }
    // reduce over the 16-lane N dimension (C layout: lane = N, vgpr = M)
#pragma unroll
    for (int v = 0; v < 8; ++v) {
      float p = part[v];
      p += __shfl_xor(p, 1, 32);
      p += __shfl_xor(p, 2, 32);
      p += __shfl_xor(p, 4, 32);
      p += __shfl_xor(p, 8, 32);
      if (mrow == 0) fpart[wave * 16 + v + ((lane >> 4) << 3)] = p;
    }
    __syncthreads();
    if (tid < 16) {
      float f = bp0;
#pragma unroll
      for (int w = 0; w < 8; ++w) f += fpart[w * 16 + tid];
      out[(size_t)(b0 + tid) * TT + t] = f;
      if (t + 1 < TT) sS[tid] = 0.5f * (f + px[(size_t)(b0 + tid) * TT + t + 1]);
    }
    __syncthreads();
  }
}

extern "C" void kernel_launch(void* const* d_in, const int* in_sizes, int n_in,
                              void* d_out, int out_size, void* d_ws, size_t ws_size,
                              hipStream_t stream) {
  const float* x     = (const float*)d_in[0];
  // d_in[1] timestamp: unused by the math
  const float* gamma = (const float*)d_in[2];
  const float* beta  = (const float*)d_in[3];
  const float* W_ih0 = (const float*)d_in[4];
  const float* b_ih0 = (const float*)d_in[5];
  // d_in[6] W_hh0: dead (h_prev == 0)
  const float* b_hh0 = (const float*)d_in[7];
  const float* W_ih1 = (const float*)d_in[8];
  const float* b_ih1 = (const float*)d_in[9];
  // d_in[10] W_hh1: dead
  const float* b_hh1 = (const float*)d_in[11];
  const float* Wp    = (const float*)d_in[12];
  const float* bp    = (const float*)d_in[13];
  float* out = (float*)d_out;

  float* ws = (float*)d_ws;
  float* sum   = ws;             // 1024
  float* sumsq = ws + 1024;      // 1024
  float* cW    = ws + 2048;      // 1024
  float* Cacc  = ws + 3072;      // 1 (+pad to 3136)
  float* px    = ws + 3136;      // 32768
  _Float16* wpack = (_Float16*)(ws + 3136 + 32768);  // 3072*1024 f16 = 6 MB

  k_zero  <<<1, 256, 0, stream>>>(ws);
  k_stats <<<512, 256, 0, stream>>>(x, sum, sumsq);
  k_params<<<4, 256, 0, stream>>>(sum, sumsq, gamma, beta, Wp, cW, Cacc);
  k_px    <<<4096, 256, 0, stream>>>(x, cW, Cacc, bp, px);
  k_pack  <<<12288, 256, 0, stream>>>(W_ih1, wpack);
  k_scan  <<<4, 256, 0, stream>>>(px, W_ih0, b_ih0, b_hh0,
                                  b_ih1, b_hh1, Wp, bp, wpack, out);
}
